// BiasAttention_71511205478974
// MI455X (gfx1250) — matrix operand order repaired and checked by reading, not compiled, not run on hardware
//
#include <hip/hip_runtime.h>
#include <hip/hip_bf16.h>

// ---------------------------------------------------------------------------
// Fused spatial-attention (BiasAttention) for MI455X / gfx1250.
// One workgroup (256 threads = 8 wave32) per (b,t) token:
//   phase0: stage x[b,t,17,512] -> LDS bf16 (padded with a shared zero row)
//   phase1: QKV GEMM via v_wmma_f32_16x16x32_bf16 (wave w computes head w cols)
//   phase2: per-head 17x17 scores (WMMA) + sd-bias softmax + AV (WMMA)
//   phase3: output projection (WMMA) + bias, fp32 store
// Weights are pre-converted fp32->bf16 into d_ws by a prologue kernel.
// ---------------------------------------------------------------------------

typedef __attribute__((ext_vector_type(16))) __bf16 bf16x16;
typedef __attribute__((ext_vector_type(8)))  __bf16 bf16x8;
typedef __attribute__((ext_vector_type(8)))  float  f32x8;

#define B_  32
#define T_  243
#define J_  17
#define C_  512
#define H_  8
#define HD  64

// ---- static LDS layout (bytes) --------------------------------------------
// qkv : 18 rows x 1544 bf16 = 55584 @ 0        (rows 0..16 data, row 17 zeros)
// attn: 18 rows x  520 bf16 = 18720 @ 55584
// D   :                             @ 74304
//   phase1 : xs 18 x 520 bf16          = 18720
//   phase2 : scores 8 x 32 x 36 f32    = 36864 @ D
//            Vt     8 x 64 x 40 bf16   = 40960 @ D + 36864
#define QKV_STRIDE 1544
#define ATT_STRIDE 520
#define ATT_OFF    55584
#define D_OFF      74304
#define SC_STRIDE  36
#define VT_OFF     (D_OFF + 36864)
#define VT_STRIDE  40
#define SMEM_BYTES (VT_OFF + 40960)   // 152128 B

// ---- fragment loaders ------------------------------------------------------
__device__ __forceinline__ bf16x16 concat8(bf16x8 lo, bf16x8 hi) {
  return __builtin_shufflevector(lo, hi, 0,1,2,3,4,5,6,7,8,9,10,11,12,13,14,15);
}

// A-matrix 16x32 bf16 fragment: lane = row (m0 + lane%16, clamped to zero row),
// 16 contiguous k at k0 + 16*(lane/16) (two 16B LDS loads).
__device__ __forceinline__ bf16x16 load_a_lds(const __bf16* base, int stride,
                                              int m0, int k0, int maxRow) {
  int lane = threadIdx.x & 31;
  int r = m0 + (lane & 15);
  if (r > maxRow) r = maxRow;                 // shared zero row
  int kb = k0 + ((lane >> 4) << 4);
  const __bf16* p = base + r * stride + kb;
  bf16x8 lo = *(const bf16x8*)p;
  bf16x8 hi = *(const bf16x8*)(p + 8);
  return concat8(lo, hi);
}

// B-matrix 32x16 bf16 fragment from row-major W[n][k]: lane = column n,
// 16 contiguous k at k0 + 16*(lane/16) (one 32B = two 16B LDS loads).
__device__ __forceinline__ bf16x16 load_b_lds(const __bf16* base, int stride,
                                              int n0, int k0, int maxRow) {
  int lane = threadIdx.x & 31;
  int r = n0 + (lane & 15);
  if (r > maxRow) r = maxRow;
  int kb = k0 + ((lane >> 4) << 4);
  const __bf16* p = base + r * stride + kb;
  bf16x8 lo = *(const bf16x8*)p;
  bf16x8 hi = *(const bf16x8*)(p + 8);
  return concat8(lo, hi);
}

__device__ __forceinline__ bf16x16 load_b_glb(const __bf16* __restrict__ base,
                                              int stride, int n0, int k0) {
  int lane = threadIdx.x & 31;
  const __bf16* p = base + (size_t)(n0 + (lane & 15)) * stride
                         + k0 + ((lane >> 4) << 4);
  bf16x8 lo = *(const bf16x8*)p;
  bf16x8 hi = *(const bf16x8*)(p + 8);
  return concat8(lo, hi);
}

__device__ __forceinline__ f32x8 wmma_bf16(bf16x16 a, bf16x16 b, f32x8 c) {
  return __builtin_amdgcn_wmma_f32_16x16x32_bf16(
      /*neg_a=*/false, a, /*neg_b=*/false, b,
      /*c_mod=*/(short)0, c, /*reuse_a=*/false, /*reuse_b=*/false);
}

// ---- prologue: fp32 -> bf16 weight conversion into workspace ---------------
__global__ __launch_bounds__(256)
void convert_weights(const float* __restrict__ wqkv,
                     const float* __restrict__ wproj,
                     __bf16* __restrict__ out) {
  const int NQ = 3 * C_ * C_;            // 786432
  const int NP = C_ * C_;                // 262144
  int i = blockIdx.x * 256 + threadIdx.x;
  if (i < NQ)            out[i]      = (__bf16)wqkv[i];
  else if (i < NQ + NP)  out[i]      = (__bf16)wproj[i - NQ];
}

// ---- fused attention kernel ------------------------------------------------
__global__ __launch_bounds__(256, 1)
void fused_bias_attn(const float*  __restrict__ x,
                     const __bf16* __restrict__ wqkv_bf,
                     const __bf16* __restrict__ wproj_bf,
                     const float*  __restrict__ bproj,
                     const float*  __restrict__ sd_emb,
                     const int*    __restrict__ sd,
                     float*        __restrict__ out) {
  __shared__ __attribute__((aligned(16))) char smem[SMEM_BYTES];
  __bf16* qkv = (__bf16*)(smem);
  __bf16* att = (__bf16*)(smem + ATT_OFF);
  __bf16* xs  = (__bf16*)(smem + D_OFF);
  float*  sc  = (float*)(smem + D_OFF);
  __bf16* vt  = (__bf16*)(smem + VT_OFF);

  const int bt   = blockIdx.x;
  const int tid  = threadIdx.x;
  const int wave = tid >> 5;
  const int lane = tid & 31;
  const int col  = lane & 15;
  const int rb   = (lane >> 4) << 3;

  // ---- phase 0: stage x -> LDS bf16; zero the shared pad rows ----
  const float* xg = x + (size_t)bt * (J_ * C_);
  for (int i = tid; i < J_ * C_; i += 256)
    xs[(i >> 9) * ATT_STRIDE + (i & 511)] = (__bf16)xg[i];
  for (int i = tid; i < C_; i += 256) {
    xs[J_ * ATT_STRIDE + i]  = (__bf16)0.f;
    att[J_ * ATT_STRIDE + i] = (__bf16)0.f;
  }
  for (int i = tid; i < 3 * C_; i += 256)
    qkv[J_ * QKV_STRIDE + i] = (__bf16)0.f;
  __syncthreads();

  // ---- phase 1: QKV GEMM. wave w computes head w's q,k,v columns ----
  for (int t12 = 0; t12 < 12; ++t12) {
    const int p  = t12 >> 2;
    const int n0 = p * C_ + wave * HD + (t12 & 3) * 16;
    f32x8 acc0 = {}, acc1 = {};
    for (int kc = 0; kc < 16; ++kc) {
      const int k0 = kc * 32;
      bf16x16 a0 = load_a_lds(xs, ATT_STRIDE, 0,  k0, J_);
      bf16x16 a1 = load_a_lds(xs, ATT_STRIDE, 16, k0, J_);
      bf16x16 b  = load_b_glb(wqkv_bf, C_, n0, k0);
      acc0 = wmma_bf16(a0, b, acc0);
      acc1 = wmma_bf16(a1, b, acc1);
    }
#pragma unroll
    for (int r = 0; r < 8; ++r) {
      int m = rb + r;
      if (m < J_)      qkv[m * QKV_STRIDE + n0 + col] = (__bf16)acc0[r];
      int m2 = 16 + rb + r;
      if (m2 < J_)     qkv[m2 * QKV_STRIDE + n0 + col] = (__bf16)acc1[r];
    }
  }
  __syncthreads();

  // ---- phase 1b: stage V transposed: Vt[h][d][j] ----
  for (int i = tid; i < H_ * HD * 32; i += 256) {
    int h = i >> 11, rem = i & 2047, d = rem >> 5, j = rem & 31;
    __bf16 v = (j < J_) ? qkv[j * QKV_STRIDE + 2 * C_ + h * HD + d]
                        : (__bf16)0.f;
    vt[(h * HD + d) * VT_STRIDE + j] = v;
  }
  __syncthreads();

  // ---- phase 2: scores = Q K^T * scale (WMMA), wave = head ----
  {
    const int h = wave, qoff = h * HD, koff = C_ + h * HD;
    f32x8 s00 = {}, s01 = {}, s10 = {}, s11 = {};
#pragma unroll
    for (int kc = 0; kc < 2; ++kc) {
      const int k0 = kc * 32;
      bf16x16 a0 = load_a_lds(qkv, QKV_STRIDE, 0,  qoff + k0, J_);
      bf16x16 a1 = load_a_lds(qkv, QKV_STRIDE, 16, qoff + k0, J_);
      bf16x16 b0 = load_b_lds(qkv, QKV_STRIDE, 0,  koff + k0, J_);
      bf16x16 b1 = load_b_lds(qkv, QKV_STRIDE, 16, koff + k0, J_);
      s00 = wmma_bf16(a0, b0, s00);  s01 = wmma_bf16(a0, b1, s01);
      s10 = wmma_bf16(a1, b0, s10);  s11 = wmma_bf16(a1, b1, s11);
    }
    float* shh = sc + h * 32 * SC_STRIDE;
#pragma unroll
    for (int r = 0; r < 8; ++r) {
      shh[(rb + r)      * SC_STRIDE + col]      = s00[r] * 0.125f;
      shh[(rb + r)      * SC_STRIDE + 16 + col] = s01[r] * 0.125f;
      shh[(16 + rb + r) * SC_STRIDE + col]      = s10[r] * 0.125f;
      shh[(16 + rb + r) * SC_STRIDE + 16 + col] = s11[r] * 0.125f;
    }
  }
  __syncthreads();

  // ---- phase 2b: softmax with graph-distance bias, row per lane ----
  {
    const int h = wave, m = lane;
    float* srow = sc + (h * 32 + m) * SC_STRIDE;
    const int mm = (m < J_) ? m : (J_ - 1);
    float v[32];
    float mx = -1e30f;
#pragma unroll
    for (int n = 0; n < 32; ++n) {
      float s = (n < J_) ? (srow[n] + sd_emb[sd[mm * J_ + n] * H_ + h])
                         : -1e30f;
      v[n] = s;
      mx = fmaxf(mx, s);
    }
    float sum = 0.f;
#pragma unroll
    for (int n = 0; n < 32; ++n) { float e = __expf(v[n] - mx); v[n] = e; sum += e; }
    const float inv = 1.f / sum;
    __bf16* prow = (__bf16*)srow;           // probs bf16 in place over scores
#pragma unroll
    for (int n = 0; n < 32; ++n) prow[n] = (__bf16)(v[n] * inv);
  }
  __syncthreads();

  // ---- phase 2c: O = P V (WMMA, K = 32 padded joints) ----
  {
    const int h = wave;
    __bf16* probs = (__bf16*)(sc + h * 32 * SC_STRIDE);     // stride 72 bf16
    bf16x16 a0 = load_a_lds(probs, SC_STRIDE * 2, 0,  0, 31);
    bf16x16 a1 = load_a_lds(probs, SC_STRIDE * 2, 16, 0, 31);
#pragma unroll
    for (int nt = 0; nt < 4; ++nt) {
      bf16x16 b = load_b_lds(vt + h * HD * VT_STRIDE, VT_STRIDE, nt * 16, 0, 63);
      f32x8 z = {};
      f32x8 o0 = wmma_bf16(a0, b, z);
      f32x8 o1 = wmma_bf16(a1, b, z);
#pragma unroll
      for (int r = 0; r < 8; ++r) {
        int m = rb + r;
        if (m < J_)  att[m * ATT_STRIDE + h * HD + nt * 16 + col] = (__bf16)o0[r];
        int m2 = 16 + rb + r;
        if (m2 < J_) att[m2 * ATT_STRIDE + h * HD + nt * 16 + col] = (__bf16)o1[r];
      }
    }
  }
  __syncthreads();

  // ---- phase 3: output projection + bias, fp32 store ----
  {
    float* og = out + (size_t)bt * (J_ * C_);
#pragma unroll
    for (int nt = 0; nt < 4; ++nt) {
      const int n0 = wave * HD + nt * 16;
      f32x8 acc0 = {}, acc1 = {};
      for (int kc = 0; kc < 16; ++kc) {
        const int k0 = kc * 32;
        bf16x16 a0 = load_a_lds(att, ATT_STRIDE, 0,  k0, J_);
        bf16x16 a1 = load_a_lds(att, ATT_STRIDE, 16, k0, J_);
        bf16x16 b  = load_b_glb(wproj_bf, C_, n0, k0);
        acc0 = wmma_bf16(a0, b, acc0);
        acc1 = wmma_bf16(a1, b, acc1);
      }
      const float bias = bproj[n0 + col];
#pragma unroll
      for (int r = 0; r < 8; ++r) {
        int m = rb + r;
        if (m < J_)  og[m * C_ + n0 + col]  = acc0[r] + bias;
        int m2 = 16 + rb + r;
        if (m2 < J_) og[m2 * C_ + n0 + col] = acc1[r] + bias;
      }
    }
  }
}

// ---------------------------------------------------------------------------
extern "C" void kernel_launch(void* const* d_in, const int* in_sizes, int n_in,
                              void* d_out, int out_size, void* d_ws, size_t ws_size,
                              hipStream_t stream) {
  const float* x     = (const float*)d_in[0];
  const float* wqkv  = (const float*)d_in[1];
  const float* wproj = (const float*)d_in[2];
  const float* bproj = (const float*)d_in[3];
  const float* sdemb = (const float*)d_in[4];
  const int*   sd    = (const int*)d_in[5];
  float* out = (float*)d_out;

  __bf16* wbuf = (__bf16*)d_ws;                 // 2.1 MB bf16 weights
  convert_weights<<<4096, 256, 0, stream>>>(wqkv, wproj, wbuf);
  fused_bias_attn<<<B_ * T_, 256, 0, stream>>>(
      x, wbuf, wbuf + 3 * C_ * C_, bproj, sdemb, sd, out);
}
